// SRVQPyworld_38242388804097
// MI455X (gfx1250) — matrix operand order repaired and checked by reading, not compiled, not run on hardware
//
#include <hip/hip_runtime.h>
#include <hip/hip_bf16.h>
#include <math.h>

typedef __attribute__((ext_vector_type(16))) _Float16 v16h;
typedef __attribute__((ext_vector_type(8)))  float    v8f;

#define TPB 256

// ---------------------------------------------------------------------------
// Elementwise converts / zero
// ---------------------------------------------------------------------------
__global__ void f32_to_f16_k(const float* __restrict__ in, _Float16* __restrict__ out, long n) {
  for (long i = (long)blockIdx.x * TPB + threadIdx.x; i < n; i += (long)gridDim.x * TPB)
    out[i] = (_Float16)in[i];
}

__global__ void zero_f32_k(float* __restrict__ p, int n) {
  int i = blockIdx.x * TPB + threadIdx.x;
  if (i < n) p[i] = 0.f;
}

// ---------------------------------------------------------------------------
// Pack fp32 row-major [M x Kreal] weights into WMMA-A f16 fragment tiles.
// Tile (mt,kt): 32 lanes x 16 halfs contiguous (512 halfs).
// A-layout (16-bit A 16x32, contiguous K to match B):
//   lanes 0-15 : VGPR v: K=2v,2v+1
//   lanes 16-31: VGPR v: K=16+2v,16+2v+1
//   M = lane % 16
// ---------------------------------------------------------------------------
__global__ void pack_a_f16_k(const float* __restrict__ W, _Float16* __restrict__ Ap,
                             int M, int Kreal, int KT) {
  int Mtiles = (M + 15) >> 4;
  long total = (long)Mtiles * KT * 512;
  for (long idx = (long)blockIdx.x * TPB + threadIdx.x; idx < total; idx += (long)gridDim.x * TPB) {
    int tile = (int)(idx >> 9), within = (int)(idx & 511);
    int l = within >> 4, h = within & 15;
    int mt = tile / KT, kt = tile % KT;
    int v = h >> 1, p = h & 1;
    int base = ((l < 16) ? 0 : 16) + 2 * v;
    int K = kt * 32 + base + p;
    int m = mt * 16 + (l & 15);
    float val = (m < M && K < Kreal) ? W[(long)m * Kreal + K] : 0.f;
    Ap[idx] = (_Float16)val;
  }
}

// ---------------------------------------------------------------------------
// im2col -> WMMA-B f16 fragment tiles for a 3x3 stride-2 pad-1 conv.
// B-layout (16-bit B, KxN): lane's 16 halfs cover contiguous K range
// [(lane<16?0:16), +16) at column n = lane%16.
// Activations are f16 in [C][B=32][H][W] layout.
// ---------------------------------------------------------------------------
__global__ void im2col_b_f16_k(const _Float16* __restrict__ act, _Float16* __restrict__ Bp,
                               int Cin, int Hin, int Win, int Hout, int Wout,
                               int Kreal, int KT, int Ntiles) {
  long N = (long)32 * Hout * Wout;
  long total = (long)Ntiles * KT * 512;
  long HW = (long)Hin * Win;
  for (long idx = (long)blockIdx.x * TPB + threadIdx.x; idx < total; idx += (long)gridDim.x * TPB) {
    int tile = (int)(idx >> 9), within = (int)(idx & 511);
    int l = within >> 4, h = within & 15;
    int nt = tile / KT, kt = tile % KT;
    long n = (long)nt * 16 + (l & 15);
    int K = kt * 32 + ((l < 16) ? 0 : 16) + h;
    _Float16 val = (_Float16)0.f;
    if (n < N && K < Kreal) {
      int b = (int)(n / (Hout * Wout));
      int rem = (int)(n % (Hout * Wout));
      int oh = rem / Wout, ow = rem % Wout;
      int cin = K / 9, r = K % 9, kh = r / 3, kw = r % 3;
      int ih = oh * 2 + kh - 1, iw = ow * 2 + kw - 1;
      if (ih >= 0 && ih < Hin && iw >= 0 && iw < Win)
        val = act[((long)cin * 32 + b) * HW + (long)ih * Win + iw];
    }
    Bp[idx] = val;
  }
}

// Pack GRU input features hs[b][t][c*2+w] (from act f16 [128][32][32][2])
// into WMMA-B tiles.  K=256 (KT=8), N=1024 (n = b*32 + t), Ntiles=64.
__global__ void pack_hs_b_f16_k(const _Float16* __restrict__ act, _Float16* __restrict__ Bp) {
  long total = 64l * 8 * 512;
  for (long idx = (long)blockIdx.x * TPB + threadIdx.x; idx < total; idx += (long)gridDim.x * TPB) {
    int tile = (int)(idx >> 9), within = (int)(idx & 511);
    int l = within >> 4, h = within & 15;
    int nt = tile / 8, kt = tile % 8;
    int n = nt * 16 + (l & 15);
    int K = kt * 32 + ((l < 16) ? 0 : 16) + h;
    int b = n >> 5, t = n & 31;
    int c = K >> 1, w = K & 1;
    Bp[idx] = act[((long)c * 32 + b) * 64 + t * 2 + w];
  }
}

// ---------------------------------------------------------------------------
// Register-blocked WMMA GEMM: each wave computes a 32x32 output block
// (2 M-tiles x 2 N-tiles, 4 f32 accumulators).  4 WMMAs per 4 fragment
// loads -> 16 FLOP/byte of global traffic, 2x the single-tile version.
// Requires Mtiles and Ntiles even (true for every GEMM in this model).
// ---------------------------------------------------------------------------
__global__ void wmma_gemm_k(const _Float16* __restrict__ Ap, const _Float16* __restrict__ Bp,
                            float* __restrict__ C, int M, int N, int KT, int Ntiles, int ldc) {
  int wave = threadIdx.x >> 5;
  int lane = threadIdx.x & 31;
  long sb = (long)blockIdx.x * 8 + wave;
  int NtB = Ntiles >> 1;
  int mtb = (int)(sb / NtB);
  int ntb = (int)(sb % NtB);
  if (mtb * 32 >= M) return;                 // wave-uniform: EXEC all-ones for WMMA

  v8f acc00 = {}, acc01 = {}, acc10 = {}, acc11 = {};
  const _Float16* a0 = Ap + ((long)(2 * mtb) * KT) * 512 + lane * 16;
  const _Float16* a1 = a0 + (long)KT * 512;
  const _Float16* b0 = Bp + ((long)(2 * ntb) * KT) * 512 + lane * 16;
  const _Float16* b1 = b0 + (long)KT * 512;

  for (int kt = 0; kt < KT; ++kt) {
    v16h A0 = *(const v16h*)a0;
    v16h A1 = *(const v16h*)a1;
    v16h B0 = *(const v16h*)b0;
    v16h B1 = *(const v16h*)b1;
    a0 += 512; a1 += 512; b0 += 512; b1 += 512;
    if (kt + 1 < KT) {
      __builtin_prefetch(a0, 0, 3); __builtin_prefetch(a1, 0, 3);
      __builtin_prefetch(b0, 0, 3); __builtin_prefetch(b1, 0, 3);
    }
    acc00 = __builtin_amdgcn_wmma_f32_16x16x32_f16(false, A0, false, B0, (short)0, acc00, false, false);
    acc01 = __builtin_amdgcn_wmma_f32_16x16x32_f16(false, A0, false, B1, (short)0, acc01, false, false);
    acc10 = __builtin_amdgcn_wmma_f32_16x16x32_f16(false, A1, false, B0, (short)0, acc10, false, false);
    acc11 = __builtin_amdgcn_wmma_f32_16x16x32_f16(false, A1, false, B1, (short)0, acc11, false, false);
  }

  // C/D layout: lane<16 -> M=r, lane>=16 -> M=8+r; N = lane%16
  int ncol = lane & 15;
  int mrow = (lane >> 4) << 3;
  {
    int n0 = (2 * ntb) * 16 + ncol, n1 = n0 + 16;
    int m0 = (2 * mtb) * 16 + mrow, m1 = m0 + 16;
#pragma unroll
    for (int r = 0; r < 8; ++r) {
      if (n0 < N) {
        C[(long)(m0 + r) * ldc + n0] = acc00[r];
        C[(long)(m1 + r) * ldc + n0] = acc10[r];
      }
      if (n1 < N) {
        C[(long)(m0 + r) * ldc + n1] = acc01[r];
        C[(long)(m1 + r) * ldc + n1] = acc11[r];
      }
    }
  }
}

// ---------------------------------------------------------------------------
// BatchNorm: per-channel sum / sumsq over raw [C][Nelem]
// ---------------------------------------------------------------------------
__global__ void bn_reduce_k(const float* __restrict__ raw, float* __restrict__ stats,
                            int C, long Nelem) {
  int c = blockIdx.y;
  const float* p = raw + (long)c * Nelem;
  float s = 0.f, s2 = 0.f;
  for (long i = (long)blockIdx.x * TPB + threadIdx.x; i < Nelem; i += (long)gridDim.x * TPB) {
    float v = p[i]; s += v; s2 += v * v;
  }
  __shared__ float ls[TPB], ls2[TPB];
  int tid = threadIdx.x;
  ls[tid] = s; ls2[tid] = s2;
  __syncthreads();
  for (int o = TPB / 2; o > 0; o >>= 1) {
    if (tid < o) { ls[tid] += ls[tid + o]; ls2[tid] += ls2[tid + o]; }
    __syncthreads();
  }
  if (tid == 0) { atomicAdd(&stats[c], ls[0]); atomicAdd(&stats[C + c], ls2[0]); }
}

__global__ void bn_finalize_k(const float* __restrict__ stats, const float* __restrict__ g,
                              const float* __restrict__ b, float* __restrict__ scale,
                              float* __restrict__ shift, int C, float invN) {
  int c = blockIdx.x * blockDim.x + threadIdx.x;
  if (c < C) {
    float mu  = stats[c] * invN;
    float var = stats[C + c] * invN - mu * mu;
    float inv = rsqrtf(var + 1e-5f);
    float sc  = g[c] * inv;
    scale[c] = sc;
    shift[c] = b[c] - mu * sc;
  }
}

__global__ void bn_relu_f16_k(const float* __restrict__ raw, const float* __restrict__ scale,
                              const float* __restrict__ shift, _Float16* __restrict__ out,
                              int C, long Nelem) {
  long total = (long)C * Nelem;
  for (long i = (long)blockIdx.x * TPB + threadIdx.x; i < total; i += (long)gridDim.x * TPB) {
    int c = (int)(i / Nelem);
    float v = raw[i] * scale[c] + shift[c];
    out[i] = (_Float16)fmaxf(v, 0.f);
  }
}

// ---------------------------------------------------------------------------
// GRU recurrence (B=32, H=64, T=32).  xp precomputed by WMMA GEMM as
// [192][1024] with n = b*32 + t.  Single block, h/hp resident in LDS.
// ---------------------------------------------------------------------------
__global__ void gru_k(const float* __restrict__ xp, const float* __restrict__ Whh,
                      const float* __restrict__ bih, const float* __restrict__ bhh,
                      float* __restrict__ hT) {
  __shared__ float h[32 * 64];
  __shared__ float hp[32 * 192];
  int tid = threadIdx.x;
  for (int e = tid; e < 2048; e += TPB) h[e] = 0.f;
  __syncthreads();
  for (int t = 0; t < 32; ++t) {
    for (int e = tid; e < 32 * 192; e += TPB) {
      int b = e / 192, g = e % 192;
      const float* wr = Whh + g * 64;
      const float* hr = h + b * 64;
      float dot = bhh[g];
#pragma unroll 8
      for (int k = 0; k < 64; ++k) dot += hr[k] * wr[k];
      hp[e] = dot;
    }
    __syncthreads();
    for (int e = tid; e < 2048; e += TPB) {
      int b = e >> 6, j = e & 63;
      int n = b * 32 + t;
      float ri = xp[(long)j * 1024 + n] + bih[j];
      float zi = xp[(long)(64 + j) * 1024 + n] + bih[64 + j];
      float ni = xp[(long)(128 + j) * 1024 + n] + bih[128 + j];
      float rh = hp[b * 192 + j];
      float zh = hp[b * 192 + 64 + j];
      float nh = hp[b * 192 + 128 + j];
      float r  = 1.f / (1.f + expf(-(ri + rh)));
      float z  = 1.f / (1.f + expf(-(zi + zh)));
      float nn = tanhf(ni + r * nh);
      h[e] = (1.f - z) * nn + z * h[e];
    }
    __syncthreads();
  }
  for (int e = tid; e < 2048; e += TPB) hT[e] = h[e];
}

// ---------------------------------------------------------------------------
// Residual VQ for all three encoders + style concat + loss + indices.
// hT3: [3][32][64] f32. out: style(32x384) | loss(1) | idx(3x16 as float).
// ---------------------------------------------------------------------------
__global__ void rvq_k(const float* __restrict__ hT3,
                      const float* __restrict__ ep1, const float* __restrict__ ep2,
                      const float* __restrict__ ee1, const float* __restrict__ ee2,
                      const float* __restrict__ ed1, const float* __restrict__ ed2,
                      float* __restrict__ out) {
  __shared__ float z[16 * 128];
  __shared__ float zq[16 * 128];
  __shared__ float dist[16 * 7];
  __shared__ int   idxs[16];
  __shared__ float lossAcc;
  int tid = threadIdx.x;  // blockDim = 128
  if (tid == 0) lossAcc = 0.f;
  __syncthreads();
  const float* e1s[3] = {ep1, ee1, ed1};
  const float* e2s[3] = {ep2, ee2, ed2};

  for (int enc = 0; enc < 3; ++enc) {
    for (int i = tid; i < 2048; i += 128) z[i] = hT3[enc * 2048 + i];
    __syncthreads();
    for (int stage = 0; stage < 2; ++stage) {
      const float* emb = stage ? e2s[enc] : e1s[enc];
      for (int e = tid; e < 112; e += 128) {
        int r = e / 7, j = e % 7;
        float d = 0.f;
        for (int k = 0; k < 128; ++k) {
          float diff = z[r * 128 + k] - emb[j * 128 + k];
          d += diff * diff;
        }
        dist[e] = d;
      }
      __syncthreads();
      if (tid < 16) {
        int best = 0; float bd = dist[tid * 7];
        for (int j = 1; j < 7; ++j) {
          float dd = dist[tid * 7 + j];
          if (dd < bd) { bd = dd; best = j; }
        }
        idxs[tid] = best;
      }
      __syncthreads();
      float lsum = 0.f;
      for (int i = tid; i < 2048; i += 128) {
        int r = i >> 7, k = i & 127;
        float q = emb[idxs[r] * 128 + k];
        float diff = q - z[i];
        lsum += diff * diff;
        zq[i] = q;
      }
      atomicAdd(&lossAcc, lsum * (1.f + 0.4f) / 2048.f);
      __syncthreads();
      if (tid == 0) {
        float cnt[7] = {0, 0, 0, 0, 0, 0, 0};
        for (int r = 0; r < 16; ++r) cnt[idxs[r]] += 1.f;
        float usage = 0.f;
        for (int j = 0; j < 7; ++j) {
          float em = cnt[j] / 16.f;
          usage -= em * logf(em + 1e-10f);
        }
        lossAcc += 0.01f * usage;
      }
      // style (forward value of zq_st == zq): (32,64) view of zq
      for (int i = tid; i < 2048; i += 128) {
        int r = i >> 7, k = i & 127;
        int b = 2 * r + (k >> 6);
        int hcol = k & 63;
        out[b * 384 + enc * 128 + stage * 64 + hcol] = zq[i];
      }
      if (stage == 0 && tid < 16) out[12289 + enc * 16 + tid] = (float)idxs[tid];
      __syncthreads();
      if (stage == 0) {
        for (int i = tid; i < 2048; i += 128) z[i] -= zq[i];
        __syncthreads();
      }
    }
  }
  __syncthreads();
  if (tid == 0) out[12288] = lossAcc;
}

// ---------------------------------------------------------------------------
// Host orchestration
// ---------------------------------------------------------------------------
static const int CIN_[6]  = {1, 32, 32, 64, 64, 128};
static const int COUT_[6] = {32, 32, 64, 64, 128, 128};
static const int HIN_[6]  = {2000, 1000, 500, 250, 125, 63};
static const int WIN_[6]  = {80, 40, 20, 10, 5, 3};
static const int HOUT_[6] = {1000, 500, 250, 125, 63, 32};
static const int WOUT_[6] = {40, 20, 10, 5, 3, 2};

static inline int nb(long total) {
  long b = (total + TPB - 1) / TPB;
  if (b > 262144) b = 262144;
  if (b < 1) b = 1;
  return (int)b;
}

extern "C" void kernel_launch(void* const* d_in, const int* in_sizes, int n_in,
                              void* d_out, int out_size, void* d_ws, size_t ws_size,
                              hipStream_t stream) {
  (void)in_sizes; (void)n_in; (void)out_size; (void)ws_size;
  // Input order: ref_embs, pitch_mels, energy_mels, then params in dict order:
  // conv_w[0..5], bn_g[0..5], bn_b[0..5], W_ih, W_hh, b_ih, b_hh,
  // emb_p1, emb_p2, emb_e1, emb_e2, emb_d1, emb_d2
  const float* inputs[3] = {(const float*)d_in[0], (const float*)d_in[1], (const float*)d_in[2]};
  const float* convW[6]; const float* bng[6]; const float* bnb[6];
  for (int i = 0; i < 6; ++i) {
    convW[i] = (const float*)d_in[3 + i];
    bng[i]   = (const float*)d_in[9 + i];
    bnb[i]   = (const float*)d_in[15 + i];
  }
  const float* W_ih = (const float*)d_in[21];
  const float* W_hh = (const float*)d_in[22];
  const float* b_ih = (const float*)d_in[23];
  const float* b_hh = (const float*)d_in[24];
  const float* emb[6];
  for (int i = 0; i < 6; ++i) emb[i] = (const float*)d_in[25 + i];

  // workspace arena
  char* base = (char*)d_ws;
  size_t off = 0;
  auto alloc = [&](size_t bytes) -> void* {
    size_t o = (off + 255) & ~(size_t)255;
    off = o + bytes;
    return (void*)(base + o);
  };

  int Kreal[6], KT[6], Mt[6];
  _Float16* Apack[6];
  for (int i = 0; i < 6; ++i) {
    Kreal[i] = CIN_[i] * 9;
    KT[i]    = (Kreal[i] + 31) / 32;
    Mt[i]    = COUT_[i] / 16;
    Apack[i] = (_Float16*)alloc((size_t)Mt[i] * KT[i] * 512 * sizeof(_Float16));
  }
  _Float16* Aih   = (_Float16*)alloc((size_t)12 * 8 * 512 * sizeof(_Float16));
  _Float16* actA  = (_Float16*)alloc(41000000ull * sizeof(_Float16));
  _Float16* actB  = (_Float16*)alloc(41000000ull * sizeof(_Float16));
  _Float16* Bpack = (_Float16*)alloc(92400000ull * sizeof(_Float16));
  float* raw   = (float*)alloc(41000000ull * sizeof(float));
  float* stats = (float*)alloc(256 * sizeof(float));
  float* scale = (float*)alloc(128 * sizeof(float));
  float* shift = (float*)alloc(128 * sizeof(float));
  float* hT3   = (float*)alloc(3 * 2048 * sizeof(float));

  // pack all weights into WMMA-A fragment tiles (every call; deterministic)
  for (int i = 0; i < 6; ++i) {
    long tot = (long)Mt[i] * KT[i] * 512;
    pack_a_f16_k<<<nb(tot), TPB, 0, stream>>>(convW[i], Apack[i], COUT_[i], Kreal[i], KT[i]);
  }
  pack_a_f16_k<<<nb(12l * 8 * 512), TPB, 0, stream>>>(W_ih, Aih, 192, 256, 8);

  for (int enc = 0; enc < 3; ++enc) {
    f32_to_f16_k<<<nb(5120000), TPB, 0, stream>>>(inputs[enc], actA, 5120000);
    const _Float16* ain = actA;
    _Float16* aout = actB;
    for (int li = 0; li < 6; ++li) {
      int C = COUT_[li];
      long N = (long)32 * HOUT_[li] * WOUT_[li];
      int Ntiles = (int)((N + 15) / 16);
      if (Ntiles & 1) Ntiles++;   // keep even for 2x2 register blocking
      long totB = (long)Ntiles * KT[li] * 512;
      im2col_b_f16_k<<<nb(totB), TPB, 0, stream>>>(ain, Bpack, CIN_[li], HIN_[li], WIN_[li],
                                                   HOUT_[li], WOUT_[li], Kreal[li], KT[li], Ntiles);
      long superTiles = (long)(Mt[li] / 2) * (Ntiles / 2);
      wmma_gemm_k<<<(int)((superTiles + 7) / 8), TPB, 0, stream>>>(Apack[li], Bpack, raw,
                                                                   C, (int)N, KT[li], Ntiles, (int)N);
      zero_f32_k<<<1, TPB, 0, stream>>>(stats, 2 * C);
      long rb = (N + TPB - 1) / TPB; if (rb > 1024) rb = 1024;
      dim3 rg((unsigned)rb, (unsigned)C);
      bn_reduce_k<<<rg, TPB, 0, stream>>>(raw, stats, C, N);
      bn_finalize_k<<<1, 128, 0, stream>>>(stats, bng[li], bnb[li], scale, shift, C, 1.f / (float)N);
      bn_relu_f16_k<<<nb((long)C * N), TPB, 0, stream>>>(raw, scale, shift, aout, C, N);
      _Float16* tmp = aout; aout = (_Float16*)ain; ain = tmp;
    }
    // GRU input projection: xp[192][1024] = W_ih * hs^T  (WMMA)
    pack_hs_b_f16_k<<<nb(64l * 8 * 512), TPB, 0, stream>>>(ain, Bpack);
    {
      long superTiles = (12 / 2) * (64 / 2);
      wmma_gemm_k<<<(int)((superTiles + 7) / 8), TPB, 0, stream>>>(Aih, Bpack, raw,
                                                                   192, 1024, 8, 64, 1024);
    }
    gru_k<<<1, TPB, 0, stream>>>(raw, W_hh, b_ih, b_hh, hT3 + enc * 2048);
  }

  rvq_k<<<1, 128, 0, stream>>>(hT3, emb[0], emb[1], emb[2], emb[3], emb[4], emb[5], (float*)d_out);
}
